// ObservationActionTransformer_45655502356627
// MI455X (gfx1250) — matrix operand order repaired and checked by reading, not compiled, not run on hardware
//
#include <hip/hip_runtime.h>
#include <hip/hip_bf16.h>

// ---------------- model dims ----------------
#define OBS_ 17
#define ACT_ 6
#define DMODEL 20
#define NHEAD 4
#define DHEAD 5
#define SEQ 3
#define FFD 128

// ---------------- LDS layout (halfs) ----------------
#define OW1    0            // 112(+pad read) x 32
#define OW2    3584         // 304(+pad read) x 128
#define OW3    42496        // 32 x 320
#define OWOBS  52736        // 32 x 32
#define OWACT  53760        // 32 x 32 (cols 17..22)
#define OWQKV  54784        // 3 x 64 x 32
#define OWO    60928        // 3 x 32 x 32
#define OWF1   64000        // 3 x 128 x 32
#define OWF2   76288        // 3 x 32 x 128
#define WTOT_H 88576
// per-wave scratch (halfs, relative)
#define SOA    0            // 16 x 32
#define SSEQ   512          // 48 x 32
#define SH1    512          // 16 x 128 (aliases SSEQ, disjoint in time)
#define SQKV   2048         // 48 x 64  (encoder phase)
#define SH2    2560         // 16 x 320 (MLP phase)
#define SFFH   2048         // 16 x 128 (FFN phase, per position)
#define STMP2  4096         // 48 x 32  (FFN out)
#define SATT   5120         // 48 x 32
#define STMP   6656         // 48 x 32  (attn out-proj)
#define SSTG   8192         // TDM staging: 16x17 f32 obs (544h) + 16x6 f32 act (192h)
#define SSTGA  (SSTG + 544)
#define SCR_H  8960
#define NWAVE  8
#define SCR0_H WTOT_H
// float (bias) region, offsets in floats
#define FBASE_BYTES ((WTOT_H + NWAVE * SCR_H) * 2)
#define FB1    0
#define FB2    112
#define FBM    416          // 3x32: b3+pos0, bobs+pos1, bact+pos2
#define FBQKV  512          // 3x64
#define FBO    704          // 3x32
#define FBF1   800          // 3x128
#define FBF2   1184         // 3x32
#define FG1    1280
#define FBE1   1376
#define FG2    1472
#define FBE2   1568
#define FWFC   1664
#define FBFC   1728
#define FTOT   1732
#define SMEM_BYTES (FBASE_BYTES + FTOT * 4)

typedef __attribute__((ext_vector_type(16))) _Float16 v16h;
typedef __attribute__((ext_vector_type(8)))  float    v8f;
typedef __attribute__((ext_vector_type(4)))  unsigned int v4u;
typedef __attribute__((ext_vector_type(8)))  int          v8i;
typedef __attribute__((ext_vector_type(4)))  int          v4i;

struct Params {
    const float *obs, *act;
    const float *W1, *b1, *W2, *b2, *W3, *b3;
    const float *Wobs, *bobs, *Wact, *bact, *pos;
    const float *Wqkv, *bqkv, *Wo, *bo, *g1, *be1;
    const float *Wf1, *bf1, *Wf2, *bf2, *g2, *be2;
    const float *Wfc, *bfc;
    float *out;
    int nrows;
};

__device__ __forceinline__ void wave_lds_fence() {
    asm volatile("s_wait_dscnt 0" ::: "memory");
}

// TDM: async DMA of a (rows=16) x (width f32) 2-D tile from global -> LDS (packed).
// Wave-level op (EXEC ignored); completion tracked by TENSORcnt.
__device__ __forceinline__ void tdm_load_2d(unsigned int lds_byte_off,
                                            const float* gptr,
                                            unsigned int width,
                                            unsigned int rows_avail) {
    unsigned long long ga = (unsigned long long)(const void*)gptr;
    v4u g0;
    g0[0] = 1u;                                   // count=1, user mode
    g0[1] = lds_byte_off;                         // lds_addr (bytes)
    g0[2] = (unsigned int)ga;                     // global_addr[31:0]
    g0[3] = (unsigned int)((ga >> 32) & 0x1FFFFFFu) | (2u << 30); // addr[56:32] | type=2
    v8i g1;
    g1[0] = (int)(2u << 16);                      // wg_mask=0, data_size=4B
    g1[1] = (int)(width << 16);                   // tensor_dim0[15:0] -> [31:16]
    g1[2] = (int)((width >> 16) & 0xFFFFu) | (int)((rows_avail & 0xFFFFu) << 16);
    g1[3] = (int)((rows_avail >> 16) & 0xFFFFu) | (int)(width << 16); // tile_dim0=width
    g1[4] = 16;                                   // tile_dim1=16 rows, tile_dim2=0
    g1[5] = (int)width;                           // tensor_dim0_stride[31:0]
    g1[6] = 0;
    g1[7] = 0;
    v4i z4 = {0, 0, 0, 0};
#if defined(__clang_major__) && (__clang_major__ >= 23)
    v8i z8 = {0, 0, 0, 0, 0, 0, 0, 0};
    __builtin_amdgcn_tensor_load_to_lds(g0, g1, z4, z4, z8, 0);
#else
    __builtin_amdgcn_tensor_load_to_lds(g0, g1, z4, z4, 0);
#endif
}

// copy W[out][in] (f32 global) -> f16 LDS, row stride kp, column offset coff.
__device__ __forceinline__ void fill_w(_Float16* dst, const float* src,
                                       int nout, int kin, int kp, int coff,
                                       int tid, int nthr) {
    for (int i = tid; i < nout * kin; i += nthr) {
        int r = i / kin, c = i - r * kin;
        dst[r * kp + c + coff] = (_Float16)src[i];
    }
}
__device__ __forceinline__ void copy_f(float* dst, const float* src, int n,
                                       int tid, int nthr) {
    for (int i = tid; i < n; i += nthr) dst[i] = src[i];
}

// D(16x32,f32) = bias + sum_kt A(16x32) x B(32x32); f16 store (+relu).
// Two N-tiles per iteration: shared A fragment, independent accumulators.
// nt_count must be even. Bias pre-loaded into WMMA C operand.
__device__ __forceinline__ void gemm16(const _Float16* A, int lda,
                                       const _Float16* W, int ldb,
                                       _Float16* O, int ldo,
                                       const float* bias,
                                       int nt_count, int kt_count,
                                       bool relu, int lane) {
    const int mn  = lane & 15;            // A row m / B col n / D col n
    const int kga = (lane >> 4) << 3;     // unused (0 or 8); A loads use kgb, as B does
    const int kgb = (lane >> 4) << 4;     // B K offset (0 or 16)
    const int mh  = (lane >> 4) << 3;     // D row offset (0 or 8)
    const _Float16* arow = A + mn * lda;
    for (int nt = 0; nt < nt_count; nt += 2) {
        float bv0 = bias[nt * 16 + mn];
        float bv1 = bias[nt * 16 + 16 + mn];
        v8f acc0, acc1;
#pragma unroll
        for (int r = 0; r < 8; ++r) { acc0[r] = bv0; acc1[r] = bv1; }
        const _Float16* w0 = W + (nt * 16 + mn) * ldb;
        const _Float16* w1 = w0 + 16 * ldb;
        for (int kt = 0; kt < kt_count; ++kt) {
            union { v16h v; uint32_t u[8]; } a, b0, b1;
            const uint32_t* ap  = (const uint32_t*)(arow + kt * 32);
            const uint32_t* bp0 = (const uint32_t*)(w0 + kt * 32 + kgb);
            const uint32_t* bp1 = (const uint32_t*)(w1 + kt * 32 + kgb);
#pragma unroll
            for (int r = 0; r < 8; ++r) {
                int k0 = kgb + (r << 1);
                a.u[r] = ap[k0 >> 1];
            }
#pragma unroll
            for (int r = 0; r < 8; ++r) { b0.u[r] = bp0[r]; b1.u[r] = bp1[r]; }
            acc0 = __builtin_amdgcn_wmma_f32_16x16x32_f16(
                false, a.v, false, b0.v, (short)0, acc0, false, false);
            acc1 = __builtin_amdgcn_wmma_f32_16x16x32_f16(
                false, a.v, false, b1.v, (short)0, acc1, false, false);
        }
#pragma unroll
        for (int r = 0; r < 8; ++r) {
            float v0 = acc0[r], v1 = acc1[r];
            if (relu) { v0 = fmaxf(v0, 0.0f); v1 = fmaxf(v1, 0.0f); }
            O[(mh + r) * ldo + nt * 16 + mn]      = (_Float16)v0;
            O[(mh + r) * ldo + nt * 16 + 16 + mn] = (_Float16)v1;
        }
    }
}

// Per-element multi-head attention: qkv (48x64 rows = s*16+m), out -> att (48x32).
__device__ __forceinline__ void attention(const _Float16* qkv, _Float16* att, int lane) {
    const int m = lane & 15;
    const float scale = 0.44721359549995794f;  // 1/sqrt(5)
#pragma unroll
    for (int hh = 0; hh < 2; ++hh) {
        int h = ((lane >> 4) << 1) + hh;
        float q[SEQ][DHEAD], k[SEQ][DHEAD], v[SEQ][DHEAD];
#pragma unroll
        for (int s = 0; s < SEQ; ++s) {
            const _Float16* row = qkv + (s * 16 + m) * 64 + h * DHEAD;
#pragma unroll
            for (int d = 0; d < DHEAD; ++d) {
                q[s][d] = (float)row[d];
                k[s][d] = (float)row[20 + d];
                v[s][d] = (float)row[40 + d];
            }
        }
#pragma unroll
        for (int s = 0; s < SEQ; ++s) {
            float sc[SEQ];
#pragma unroll
            for (int t = 0; t < SEQ; ++t) {
                float dsum = 0.f;
#pragma unroll
                for (int d = 0; d < DHEAD; ++d) dsum += q[s][d] * k[t][d];
                sc[t] = dsum * scale;
            }
            float mx = fmaxf(sc[0], fmaxf(sc[1], sc[2]));
            float e0 = __expf(sc[0] - mx), e1 = __expf(sc[1] - mx), e2 = __expf(sc[2] - mx);
            float inv = 1.0f / (e0 + e1 + e2);
            _Float16* orow = att + (s * 16 + m) * 32 + h * DHEAD;
#pragma unroll
            for (int d = 0; d < DHEAD; ++d)
                orow[d] = (_Float16)((e0 * v[0][d] + e1 * v[1][d] + e2 * v[2][d]) * inv);
        }
    }
}

// x = LN(x + y) over 20 cols; rows 0..47, stride 32.
__device__ __forceinline__ void ln_residual(_Float16* x, const _Float16* y,
                                            const float* g, const float* be, int lane) {
    for (int r = lane; r < 48; r += 32) {
        float v[DMODEL];
        float mean = 0.f;
#pragma unroll
        for (int c = 0; c < DMODEL; ++c) {
            v[c] = (float)x[r * 32 + c] + (float)y[r * 32 + c];
            mean += v[c];
        }
        mean *= (1.0f / DMODEL);
        float var = 0.f;
#pragma unroll
        for (int c = 0; c < DMODEL; ++c) {
            float d = v[c] - mean;
            var += d * d;
        }
        var *= (1.0f / DMODEL);
        float rstd = rsqrtf(var + 1e-5f);
#pragma unroll
        for (int c = 0; c < DMODEL; ++c)
            x[r * 32 + c] = (_Float16)((v[c] - mean) * rstd * g[c] + be[c]);
    }
}

__global__ void __launch_bounds__(256, 1)
oat_transformer_kernel(Params p) {
    extern __shared__ __align__(16) char smem[];
    _Float16* sh = (_Float16*)smem;
    float*    fb = (float*)(smem + FBASE_BYTES);
    const int tid = threadIdx.x;
    const int NT = 256;

    // ---- stage weights: f32 global -> zero-padded f16 LDS ----
    for (int i = tid; i < WTOT_H; i += NT) sh[i] = (_Float16)0.0f;
    for (int i = tid; i < FTOT; i += NT) fb[i] = 0.0f;
    __syncthreads();
    fill_w(sh + OW1,   p.W1,   100, 23,  32,  0, tid, NT);
    fill_w(sh + OW2,   p.W2,   300, 100, 128, 0, tid, NT);
    fill_w(sh + OW3,   p.W3,   20,  300, 320, 0, tid, NT);
    fill_w(sh + OWOBS, p.Wobs, 20,  17,  32,  0, tid, NT);
    fill_w(sh + OWACT, p.Wact, 20,  6,   32,  17, tid, NT);
    for (int l = 0; l < 3; ++l) {
        fill_w(sh + OWQKV + l * 2048, p.Wqkv + l * 1200, 60, 20, 32, 0, tid, NT);
        fill_w(sh + OWO   + l * 1024, p.Wo   + l * 400,  20, 20, 32, 0, tid, NT);
        fill_w(sh + OWF1  + l * 4096, p.Wf1  + l * 2560, 128, 20, 32, 0, tid, NT);
        fill_w(sh + OWF2  + l * 4096, p.Wf2  + l * 2560, 20, 128, 128, 0, tid, NT);
    }
    copy_f(fb + FB1, p.b1, 100, tid, NT);
    copy_f(fb + FB2, p.b2, 300, tid, NT);
    for (int i = tid; i < DMODEL; i += NT) {
        fb[FBM + i]      = p.b3[i]   + p.pos[i];
        fb[FBM + 32 + i] = p.bobs[i] + p.pos[20 + i];
        fb[FBM + 64 + i] = p.bact[i] + p.pos[40 + i];
    }
    for (int l = 0; l < 3; ++l) {
        copy_f(fb + FBQKV + l * 64,  p.bqkv + l * 60, 60, tid, NT);
        copy_f(fb + FBO   + l * 32,  p.bo  + l * 20, 20, tid, NT);
        copy_f(fb + FBF1  + l * 128, p.bf1 + l * 128, 128, tid, NT);
        copy_f(fb + FBF2  + l * 32,  p.bf2 + l * 20, 20, tid, NT);
        copy_f(fb + FG1   + l * 32,  p.g1  + l * 20, 20, tid, NT);
        copy_f(fb + FBE1  + l * 32,  p.be1 + l * 20, 20, tid, NT);
        copy_f(fb + FG2   + l * 32,  p.g2  + l * 20, 20, tid, NT);
        copy_f(fb + FBE2  + l * 32,  p.be2 + l * 20, 20, tid, NT);
    }
    copy_f(fb + FWFC, p.Wfc, 60, tid, NT);
    if (tid == 0) fb[FBFC] = p.bfc[0];
    __syncthreads();

    // ---- per-wave fused pipeline over tiles of 16 batch rows ----
    const int wave = tid >> 5;
    const int lane = tid & 31;
    _Float16* ws = sh + SCR0_H + wave * SCR_H;
    const unsigned int stg_byte = (unsigned int)(SCR0_H + wave * SCR_H) * 2u;
    for (int i = lane; i < SCR_H; i += 32) ws[i] = (_Float16)0.0f;
    wave_lds_fence();

    // TDM prologue: DMA tile 0 inputs into LDS staging (async, TENSORcnt)
    {
        int r0 = blockIdx.x * 256 + wave * 16;
        if (r0 + 16 <= p.nrows) {
            tdm_load_2d(stg_byte + SSTG * 2u, p.obs + (size_t)r0 * OBS_,
                        OBS_, (unsigned int)(p.nrows - r0));
            tdm_load_2d(stg_byte + SSTGA * 2u, p.act + (size_t)r0 * ACT_,
                        ACT_, (unsigned int)(p.nrows - r0));
        }
    }

    for (int it = 0; it < 2; ++it) {
        int row0 = blockIdx.x * 256 + it * 128 + wave * 16;
        if (row0 + 16 > p.nrows) continue;
        // wait for staged inputs, convert f32 staging -> f16 oa tile
        __builtin_amdgcn_s_wait_tensorcnt(0);
        const float* stg_obs = (const float*)(ws + SSTG);
        const float* stg_act = (const float*)(ws + SSTGA);
        for (int i = lane; i < 16 * OBS_; i += 32) {
            int r = i / OBS_, c = i - r * OBS_;
            ws[SOA + r * 32 + c] = (_Float16)stg_obs[i];
        }
        for (int i = lane; i < 16 * ACT_; i += 32) {
            int r = i / ACT_, c = i - r * ACT_;
            ws[SOA + r * 32 + 17 + c] = (_Float16)stg_act[i];
        }
        wave_lds_fence();  // staging fully consumed before re-issuing TDM
        if (it == 0) {
            int rn = row0 + 128;
            if (rn + 16 <= p.nrows) {
                tdm_load_2d(stg_byte + SSTG * 2u, p.obs + (size_t)rn * OBS_,
                            OBS_, (unsigned int)(p.nrows - rn));
                tdm_load_2d(stg_byte + SSTGA * 2u, p.act + (size_t)rn * ACT_,
                            ACT_, (unsigned int)(p.nrows - rn));
            }
        }
        // MLP: 23->100->300->20 (NT padded even; pad cols hit zero K-cols downstream)
        gemm16(ws + SOA, 32, sh + OW1, 32, ws + SH1, 128, fb + FB1, 8, 1, true, lane);
        wave_lds_fence();
        gemm16(ws + SH1, 128, sh + OW2, 128, ws + SH2, 320, fb + FB2, 20, 4, true, lane);
        wave_lds_fence();
        gemm16(ws + SH2, 320, sh + OW3, 320, ws + SSEQ, 32, fb + FBM, 2, 10, false, lane);
        // projections (share oa tile; bias includes pos)
        gemm16(ws + SOA, 32, sh + OWOBS, 32, ws + SSEQ + 512, 32, fb + FBM + 32, 2, 1, false, lane);
        gemm16(ws + SOA, 32, sh + OWACT, 32, ws + SSEQ + 1024, 32, fb + FBM + 64, 2, 1, false, lane);
        wave_lds_fence();
        // 3 encoder layers
        for (int l = 0; l < 3; ++l) {
            for (int mt = 0; mt < 3; ++mt)
                gemm16(ws + SSEQ + mt * 512, 32, sh + OWQKV + l * 2048, 32,
                       ws + SQKV + mt * 1024, 64, fb + FBQKV + l * 64, 4, 1, false, lane);
            wave_lds_fence();
            attention(ws + SQKV, ws + SATT, lane);
            wave_lds_fence();
            for (int mt = 0; mt < 3; ++mt)
                gemm16(ws + SATT + mt * 512, 32, sh + OWO + l * 1024, 32,
                       ws + STMP + mt * 512, 32, fb + FBO + l * 32, 2, 1, false, lane);
            wave_lds_fence();
            ln_residual(ws + SSEQ, ws + STMP, fb + FG1 + l * 32, fb + FBE1 + l * 32, lane);
            wave_lds_fence();
            // FFN per sequence position (keeps scratch small -> 8 waves/WGP)
            for (int mt = 0; mt < 3; ++mt) {
                gemm16(ws + SSEQ + mt * 512, 32, sh + OWF1 + l * 4096, 32,
                       ws + SFFH, 128, fb + FBF1 + l * 128, 8, 1, true, lane);
                wave_lds_fence();
                gemm16(ws + SFFH, 128, sh + OWF2 + l * 4096, 128,
                       ws + STMP2 + mt * 512, 32, fb + FBF2 + l * 32, 2, 4, false, lane);
                wave_lds_fence();
            }
            ln_residual(ws + SSEQ, ws + STMP2, fb + FG2 + l * 32, fb + FBE2 + l * 32, lane);
            wave_lds_fence();
        }
        // final fc: flat (s*20+d) dot Wfc + bfc
        if (lane < 16) {
            float acc = fb[FBFC];
#pragma unroll
            for (int s = 0; s < SEQ; ++s)
                for (int d = 0; d < DMODEL; ++d)
                    acc += (float)ws[SSEQ + (s * 16 + lane) * 32 + d] * fb[FWFC + s * 20 + d];
            p.out[row0 + lane] = acc;
        }
        wave_lds_fence();
    }
}

extern "C" void kernel_launch(void* const* d_in, const int* in_sizes, int n_in,
                              void* d_out, int out_size, void* d_ws, size_t ws_size,
                              hipStream_t stream) {
    Params p;
    p.obs  = (const float*)d_in[0];   p.act  = (const float*)d_in[1];
    p.W1   = (const float*)d_in[2];   p.b1   = (const float*)d_in[3];
    p.W2   = (const float*)d_in[4];   p.b2   = (const float*)d_in[5];
    p.W3   = (const float*)d_in[6];   p.b3   = (const float*)d_in[7];
    p.Wobs = (const float*)d_in[8];   p.bobs = (const float*)d_in[9];
    p.Wact = (const float*)d_in[10];  p.bact = (const float*)d_in[11];
    p.pos  = (const float*)d_in[12];
    p.Wqkv = (const float*)d_in[13];  p.bqkv = (const float*)d_in[14];
    p.Wo   = (const float*)d_in[15];  p.bo   = (const float*)d_in[16];
    p.g1   = (const float*)d_in[17];  p.be1  = (const float*)d_in[18];
    p.Wf1  = (const float*)d_in[19];  p.bf1  = (const float*)d_in[20];
    p.Wf2  = (const float*)d_in[21];  p.bf2  = (const float*)d_in[22];
    p.g2   = (const float*)d_in[23];  p.be2  = (const float*)d_in[24];
    p.Wfc  = (const float*)d_in[25];  p.bfc  = (const float*)d_in[26];
    p.out  = (float*)d_out;
    p.nrows = in_sizes[0] / OBS_;

    (void)d_ws; (void)ws_size; (void)out_size; (void)n_in;
    hipFuncSetAttribute((const void*)oat_transformer_kernel,
                        hipFuncAttributeMaxDynamicSharedMemorySize, (int)SMEM_BYTES);
    int grid = (p.nrows + 255) / 256;
    hipLaunchKernelGGL(oat_transformer_kernel, dim3(grid), dim3(256),
                       SMEM_BYTES, stream, p);
}
